// LocallyDense_54786602828589
// MI455X (gfx1250) — matrix operand order repaired and checked by reading, not compiled, not run on hardware
//
#include <hip/hip_runtime.h>

// Problem constants (match reference)
#define B_DIM    256
#define D_IN     65536
#define N_GROUPS 41
#define G_DIM    1536
#define D_OUT    256
#define BN_EPS   1e-3f
#define ALPHA    0.3f

// Tiling: 128x128 block tile, K staged in chunks of 64 (2 WMMA k-steps)
#define BM   128
#define BN   128
#define KT   64
#define LDSS (KT + 8)            // ushort row stride (144 B = 9*16, keeps b128 alignment)

typedef __attribute__((ext_vector_type(16))) __bf16        v16bf;
typedef __attribute__((ext_vector_type(8)))  float         v8f;
typedef __attribute__((ext_vector_type(4)))  unsigned int  v4u;

union FragU { v4u q[2]; v16bf v; };
static_assert(sizeof(v16bf) == 32, "v16bf must be 32 bytes");

// Pack two f32 -> packed bf16x2 with one round-add each + one v_perm_b32.
// dst[15:0] = bf16(a), dst[31:16] = bf16(b)
__device__ __forceinline__ unsigned int pack_bf16x2(float a, float b) {
    unsigned int ua = __float_as_uint(a) + 0x8000u;  // round-half-up (~RNE, inputs well-scaled)
    unsigned int ub = __float_as_uint(b) + 0x8000u;
    return __builtin_amdgcn_perm(ub, ua, 0x07060302u); // bytes: [ub.3 ub.2 ua.3 ua.2]
}

__global__ __launch_bounds__(256, 2)
void grouped_dense_wmma(const float* __restrict__ x,
                        const int*   __restrict__ gidx,
                        const float* __restrict__ W,
                        const float* __restrict__ bias,
                        const float* __restrict__ gamma,
                        const float* __restrict__ beta,
                        const float* __restrict__ mmean,
                        const float* __restrict__ mvar,
                        float*       __restrict__ out) {
    __shared__ __align__(16) unsigned short aS[BM * LDSS]; // A: [row][k]  bf16
    __shared__ __align__(16) unsigned short bS[BN * LDSS]; // B: [col][k]  bf16 (transposed)

    const int tid  = threadIdx.x;
    const int lane = tid & 31;
    const int w    = tid >> 5;                    // wave id 0..7
    const int n    = blockIdx.y;                  // group
    const int m0   = (blockIdx.x & 1) * BM;       // batch-row tile base
    const int n0   = (blockIdx.x >> 1) * BN;      // out-col tile base

    // ---- cooperative staging coordinates (pair-packed along k) ----
    // A: thread owns k-pair kpA (fixed), rows (tid>>5) ... rows = mA + 8*i
    const int kpA = (tid & 31) * 2;               // k of first elem of pair
    const int mA  = tid >> 5;                     // 0..7
    // B: thread owns column nnB (fixed), k-pairs (tid>>7) + 2*i
    const int nnB  = tid & 127;
    const int kpB  = tid >> 7;                    // 0 or 1

    // ---- wave compute layout: 2 M-tiles x 4 N-tiles of 16x16 ----
    const int half = (lane >> 4) & 1;
    const int l16  = lane & 15;
    const int mW   = (w & 3) * 32;                // wave M base (2 tiles: +0, +16)
    const int nW   = (w >> 2) * 64;               // wave N base (4 tiles)

    v8f acc[2][4] = {};

    for (int k0 = 0; k0 < G_DIM; k0 += KT) {
        __syncthreads();   // previous chunk's fragments fully consumed

        // ---- stage A: gathered x, packed f32x2 -> bf16x2 ----
        {
            const int* gp = gidx + n * G_DIM + k0 + kpA;
            const int c0 = gp[0];
            const int c1 = gp[1];
            #pragma unroll 8
            for (int i = 0; i < BM / 16; ++i) {   // 8 rows per thread
                const int m = mA + 8 * i;
                const float a0 = x[(size_t)(m0 + m) * D_IN + c0];
                const float a1 = x[(size_t)(m0 + m) * D_IN + c1];
                *(unsigned int*)&aS[m * LDSS + kpA] = pack_bf16x2(a0, a1);
            }
        }
        // ---- stage B: W coalesced over o, transposed to [col][k], packed ----
        {
            const float* wp = W + ((size_t)n * G_DIM + k0) * D_OUT + n0 + nnB;
            #pragma unroll 8
            for (int i = 0; i < 16; ++i) {        // 16 k-pairs per thread
                const int k = 2 * (kpB + 2 * i);
                const float b0v = wp[(size_t)k * D_OUT];
                const float b1v = wp[(size_t)(k + 1) * D_OUT];
                *(unsigned int*)&bS[nnB * LDSS + k] = pack_bf16x2(b0v, b1v);
            }
        }
        __syncthreads();   // tiles staged

        // ---- 2 WMMA k-steps of 32; 8 WMMAs per step with frag reuse ----
        #pragma unroll
        for (int kk = 0; kk < KT / 32; ++kk) {
            const int ks = kk * 32;
            FragU a[2], bf[4];
            const int akb = ks + (half ? 8 : 0);
            #pragma unroll
            for (int t = 0; t < 2; ++t) {
                const int row = mW + t * 16 + l16;
                a[t].q[0] = *(const v4u*)&aS[row * LDSS + akb];
                a[t].q[1] = *(const v4u*)&aS[row * LDSS + akb + 16];
            }
            const int bkb = ks + (half ? 8 : 0);
            #pragma unroll
            for (int j = 0; j < 4; ++j) {
                const int col = nW + j * 16 + l16;
                bf[j].q[0] = *(const v4u*)&bS[col * LDSS + bkb];
                bf[j].q[1] = *(const v4u*)&bS[col * LDSS + bkb + 16];
            }
            #pragma unroll
            for (int t = 0; t < 2; ++t) {
                #pragma unroll
                for (int j = 0; j < 4; ++j) {
                    acc[t][j] = __builtin_amdgcn_wmma_f32_16x16x32_bf16(
                        false, a[t].v, false, bf[j].v, (short)0, acc[t][j], false, false);
                }
            }
        }
    }

    // ---- epilogue: bias + LeakyReLU(0.3) + inference BN, store f32 ----
    // C/D layout: VGPR r -> row r (lanes 0-15) or 8+r (lanes 16-31); lane%16 -> column
    #pragma unroll
    for (int j = 0; j < 4; ++j) {
        const int col   = n0 + nW + j * 16 + l16;
        const float bsv = bias[n * D_OUT + col];
        const float inv = gamma[col] * rsqrtf(mvar[col] + BN_EPS);
        const float sh  = beta[col] - mmean[col] * inv;
        #pragma unroll
        for (int t = 0; t < 2; ++t) {
            const int rbase = m0 + mW + t * 16 + (half ? 8 : 0);
            #pragma unroll
            for (int r = 0; r < 8; ++r) {
                float v = acc[t][j][r] + bsv;
                v = (v >= 0.0f) ? v : ALPHA * v;
                v = v * inv + sh;
                out[((size_t)(rbase + r) * N_GROUPS + n) * D_OUT + col] = v;
            }
        }
    }
}

extern "C" void kernel_launch(void* const* d_in, const int* in_sizes, int n_in,
                              void* d_out, int out_size, void* d_ws, size_t ws_size,
                              hipStream_t stream) {
    (void)in_sizes; (void)n_in; (void)out_size; (void)d_ws; (void)ws_size;
    const float* x     = (const float*)d_in[0];
    const int*   gidx  = (const int*)  d_in[1];
    const float* W     = (const float*)d_in[2];
    const float* bias  = (const float*)d_in[3];
    const float* gamma = (const float*)d_in[4];
    const float* beta  = (const float*)d_in[5];
    const float* mmean = (const float*)d_in[6];
    const float* mvar  = (const float*)d_in[7];
    float* out = (float*)d_out;

    dim3 grid(4, N_GROUPS);    // 2x2 tiles of 128x128 per group, 164 blocks total
    dim3 block(256);           // 8 wave32 waves
    grouped_dense_wmma<<<grid, block, 0, stream>>>(x, gidx, W, bias, gamma, beta,
                                                   mmean, mvar, out);
}
